// ImprovedGCN_30356828848495
// MI455X (gfx1250) — matrix operand order repaired and checked by reading, not compiled, not run on hardware
//
#include <hip/hip_runtime.h>

// ---------------------------------------------------------------------------
// Types for CDNA5 WMMA (wave32): V_WMMA_F32_16X16X32_BF16
// ---------------------------------------------------------------------------
typedef __attribute__((ext_vector_type(16))) __bf16 v16bf;
typedef __attribute__((ext_vector_type(8)))  float  v8f;

union Frag16 {
    uint4 q[2];   // 32 bytes = 16 bf16
    v16bf v;
};

#define EPS 1e-5f

// round-to-nearest-even f32 -> bf16 (raw u16)
__device__ __forceinline__ unsigned short f2bf(float f) {
    union { float f; unsigned int u; } c; c.f = f;
    unsigned int r = c.u + 0x7FFFu + ((c.u >> 16) & 1u);
    return (unsigned short)(r >> 16);
}

// ---------------------------------------------------------------------------
// GEMM: C[M,Nc] (f32) = A[M,K] (bf16 row-major) * Bt[Nc,K] (bf16, B transposed)
// Register-blocked: each wave computes a 2(M) x 4(N) grid of 16x16 WMMA tiles
// (32 rows x 64 cols). 8 waves/block cover 512 output columns.
// A-frag reused across 4 WMMAs, B-frag across 2: 12 b128 loads per 8 WMMAs.
// M%32==0, Nc%64==0, K%32==0 guaranteed by the network dims.
// ---------------------------------------------------------------------------
__global__ void gemm_bf16_wmma(const unsigned short* __restrict__ A,
                               const unsigned short* __restrict__ Bt,
                               float* __restrict__ C,
                               int M, int Nc, int K) {
    const int wave = threadIdx.x >> 5;
    const int lane = threadIdx.x & 31;
    const int half = lane >> 4;   // which half of the wave
    const int lr   = lane & 15;   // row (A) / col (B,C) within tile
    const int m0 = blockIdx.x * 32;
    const int n0 = blockIdx.y * 512 + wave * 64;
    if (n0 >= Nc) return;         // wave-uniform: EXEC all-ones for WMMA

    const unsigned short* Ar0 = A + (size_t)(m0 + lr) * K;
    const unsigned short* Ar1 = Ar0 + (size_t)16 * K;
    const unsigned short* Br  = Bt + (size_t)(n0 + lr) * K;
    const size_t bstep = (size_t)16 * K;  // next N-tile's B row

    const int ka = half * 8;      // A: lanes 0-15 K=0..7/16..23, lanes 16-31 K=8..15/24..31
    const int kb = half * 8;      // B: lanes 0-15 K=0..7/16..23, lanes 16-31 K=8..15/24..31

    v8f acc[2][4] = {};
    for (int k0 = 0; k0 < K; k0 += 32) {
        Frag16 a[2], b[4];
        a[0].q[0] = *(const uint4*)(Ar0 + k0 + ka);
        a[0].q[1] = *(const uint4*)(Ar0 + k0 + ka + 16);
        a[1].q[0] = *(const uint4*)(Ar1 + k0 + ka);
        a[1].q[1] = *(const uint4*)(Ar1 + k0 + ka + 16);
        #pragma unroll
        for (int j = 0; j < 4; ++j) {
            const unsigned short* br = Br + (size_t)j * bstep;
            b[j].q[0] = *(const uint4*)(br + k0 + kb);
            b[j].q[1] = *(const uint4*)(br + k0 + kb + 16);
        }
        #pragma unroll
        for (int i = 0; i < 2; ++i)
            #pragma unroll
            for (int j = 0; j < 4; ++j)
                acc[i][j] = __builtin_amdgcn_wmma_f32_16x16x32_bf16(
                                false, a[i].v, false, b[j].v, (short)0,
                                acc[i][j], false, false);
    }

    // C/D layout: VGPR r -> row half*8 + r, lane%16 -> column
    #pragma unroll
    for (int i = 0; i < 2; ++i) {
        #pragma unroll
        for (int j = 0; j < 4; ++j) {
            float* Cp = C + (size_t)(m0 + i * 16 + half * 8) * Nc + (n0 + j * 16 + lr);
            #pragma unroll
            for (int r = 0; r < 8; ++r) Cp[(size_t)r * Nc] = acc[i][j][r];
        }
    }
}

// ---------------------------------------------------------------------------
// Graph-norm prep
// ---------------------------------------------------------------------------
__global__ void k_fill(float* p, long long n, float v) {
    long long i = (long long)blockIdx.x * blockDim.x + threadIdx.x;
    if (i < n) p[i] = v;
}

__global__ void k_deg(float* deg, const int* __restrict__ dst, int E) {
    int e = blockIdx.x * blockDim.x + threadIdx.x;
    if (e < E) atomicAdd(&deg[dst[e]], 1.0f);
}

__global__ void k_dinv_sn(const float* __restrict__ deg, float* dinv, float* sn, int N) {
    int i = blockIdx.x * blockDim.x + threadIdx.x;
    if (i < N) { float d = rsqrtf(deg[i]); dinv[i] = d; sn[i] = d * d; }
}

__global__ void k_en(const float* __restrict__ dinv, const int* __restrict__ src,
                     const int* __restrict__ dst, float* en, int E) {
    int e = blockIdx.x * blockDim.x + threadIdx.x;
    if (e < E) en[e] = dinv[src[e]] * dinv[dst[e]];
}

// ---------------------------------------------------------------------------
// Conversions
// ---------------------------------------------------------------------------
__global__ void k_cvt_bf16(const float* __restrict__ in, unsigned short* __restrict__ out,
                           long long n) {
    long long i = (long long)blockIdx.x * blockDim.x + threadIdx.x;
    if (i < n) out[i] = f2bf(in[i]);
}

// W[K,Nc] f32 -> Wt[Nc,K] bf16
__global__ void k_wt(const float* __restrict__ W, unsigned short* __restrict__ Wt,
                     int K, int Nc) {
    long long i = (long long)blockIdx.x * blockDim.x + threadIdx.x;
    long long tot = (long long)K * Nc;
    if (i < tot) {
        int k = (int)(i / Nc), n = (int)(i % Nc);
        Wt[(size_t)n * K + k] = f2bf(W[i]);
    }
}

// ---------------------------------------------------------------------------
// GCN aggregation: agg = sn[i]*hw[i] + bias  then += en[e]*hw[src] scattered
// ---------------------------------------------------------------------------
__global__ void k_self_init(float* __restrict__ agg, const float* __restrict__ hw,
                            const float* __restrict__ sn, const float* __restrict__ bias,
                            long long NH, int H) {
    long long i = (long long)blockIdx.x * blockDim.x + threadIdx.x;
    if (i < NH) {
        int f = (int)(i % H);
        agg[i] = sn[i / H] * hw[i] + bias[f];
    }
}

__global__ void k_scatter(float* __restrict__ agg, const float* __restrict__ hw,
                          const float* __restrict__ en, const int* __restrict__ src,
                          const int* __restrict__ dst, int H) {
    int e = blockIdx.x;
    float w = en[e];
    const float* hs = hw + (size_t)src[e] * H;
    float* ad = agg + (size_t)dst[e] * H;
    for (int f = threadIdx.x; f < H; f += blockDim.x)
        atomicAdd(&ad[f], w * hs[f]);
}

// ---------------------------------------------------------------------------
// Fused BatchNorm(eval) + ReLU
// ---------------------------------------------------------------------------
__global__ void k_bn_relu_bf16(const float* __restrict__ agg, unsigned short* __restrict__ out,
                               const float* __restrict__ g, const float* __restrict__ be,
                               const float* __restrict__ m, const float* __restrict__ v,
                               long long NH, int H) {
    long long i = (long long)blockIdx.x * blockDim.x + threadIdx.x;
    if (i < NH) {
        int f = (int)(i % H);
        float val = (agg[i] - m[f]) * rsqrtf(v[f] + EPS) * g[f] + be[f];
        out[i] = f2bf(fmaxf(val, 0.0f));
    }
}

__global__ void k_bn_relu_f32(float* __restrict__ agg,
                              const float* __restrict__ g, const float* __restrict__ be,
                              const float* __restrict__ m, const float* __restrict__ v,
                              long long NH, int H) {
    long long i = (long long)blockIdx.x * blockDim.x + threadIdx.x;
    if (i < NH) {
        int f = (int)(i % H);
        float val = (agg[i] - m[f]) * rsqrtf(v[f] + EPS) * g[f] + be[f];
        agg[i] = fmaxf(val, 0.0f);
    }
}

// out = relu(h3 + x@Wres + bres)
__global__ void k_final(float* __restrict__ out, const float* __restrict__ h3,
                        const float* __restrict__ idr, const float* __restrict__ bres,
                        long long NH, int H) {
    long long i = (long long)blockIdx.x * blockDim.x + threadIdx.x;
    if (i < NH) {
        int f = (int)(i % H);
        out[i] = fmaxf(h3[i] + idr[i] + bres[f], 0.0f);
    }
}

// ---------------------------------------------------------------------------
// Host orchestration
// ---------------------------------------------------------------------------
static inline size_t alignUp(size_t x, size_t a) { return (x + a - 1) & ~(a - 1); }

extern "C" void kernel_launch(void* const* d_in, const int* in_sizes, int n_in,
                              void* d_out, int out_size, void* d_ws, size_t ws_size,
                              hipStream_t stream) {
    const float* x    = (const float*)d_in[0];
    const int*   ei   = (const int*)d_in[1];
    const float* W1   = (const float*)d_in[2];
    const float* b1   = (const float*)d_in[3];
    const float* W2   = (const float*)d_in[4];
    const float* b2   = (const float*)d_in[5];
    const float* W3   = (const float*)d_in[6];
    const float* b3   = (const float*)d_in[7];
    const float* g1   = (const float*)d_in[8],  *be1 = (const float*)d_in[9];
    const float* m1   = (const float*)d_in[10], *v1  = (const float*)d_in[11];
    const float* g2   = (const float*)d_in[12], *be2 = (const float*)d_in[13];
    const float* m2   = (const float*)d_in[14], *v2  = (const float*)d_in[15];
    const float* g3   = (const float*)d_in[16], *be3 = (const float*)d_in[17];
    const float* m3   = (const float*)d_in[18], *v3  = (const float*)d_in[19];
    const float* Wres = (const float*)d_in[20];
    const float* bres = (const float*)d_in[21];
    float* out = (float*)d_out;

    const int H1    = in_sizes[3];
    const int H2    = in_sizes[5];
    const int DOUT  = in_sizes[7];
    const int DIN   = in_sizes[2] / H1;
    const int N     = in_sizes[0] / DIN;
    const int E     = in_sizes[1] / 2;
    const int HMAX  = DOUT > H1 ? (DOUT > H2 ? DOUT : H2) : H1;
    const int HB    = H1 > H2 ? H1 : H2;

    const int* src = ei;
    const int* dst = ei + E;

    // --- workspace carving ---
    char* ws = (char*)d_ws; size_t off = 0;
    auto carve = [&](size_t bytes) -> void* {
        void* p = ws + off; off = alignUp(off + bytes, 256); return p;
    };
    float*          deg  = (float*)carve((size_t)N * 4);
    float*          dinv = (float*)carve((size_t)N * 4);
    float*          sn   = (float*)carve((size_t)N * 4);
    float*          en   = (float*)carve((size_t)E * 4);
    unsigned short* W1t  = (unsigned short*)carve((size_t)DIN * H1 * 2);
    unsigned short* W2t  = (unsigned short*)carve((size_t)H1 * H2 * 2);
    unsigned short* W3t  = (unsigned short*)carve((size_t)H2 * DOUT * 2);
    unsigned short* Wrt  = (unsigned short*)carve((size_t)DIN * DOUT * 2);
    unsigned short* xb   = (unsigned short*)carve((size_t)N * DIN * 2);
    unsigned short* hb   = (unsigned short*)carve((size_t)N * HB * 2);
    float*          idr  = (float*)carve((size_t)N * DOUT * 4);
    float*          hw   = (float*)carve((size_t)N * HMAX * 4);
    float*          agg  = (float*)carve((size_t)N * HMAX * 4);
    (void)ws_size;

    const int T = 256;
    auto blocks = [&](long long n) { return (unsigned)((n + T - 1) / T); };

    // --- graph norms ---
    k_fill<<<blocks(N), T, 0, stream>>>(deg, N, 1.0f);                 // self-loop
    k_deg<<<blocks(E), T, 0, stream>>>(deg, dst, E);
    k_dinv_sn<<<blocks(N), T, 0, stream>>>(deg, dinv, sn, N);
    k_en<<<blocks(E), T, 0, stream>>>(dinv, src, dst, en, E);

    // --- one-time conversions ---
    k_cvt_bf16<<<blocks((long long)N * DIN), T, 0, stream>>>(x, xb, (long long)N * DIN);
    k_wt<<<blocks((long long)DIN * H1), T, 0, stream>>>(W1, W1t, DIN, H1);
    k_wt<<<blocks((long long)H1 * H2), T, 0, stream>>>(W2, W2t, H1, H2);
    k_wt<<<blocks((long long)H2 * DOUT), T, 0, stream>>>(W3, W3t, H2, DOUT);
    k_wt<<<blocks((long long)DIN * DOUT), T, 0, stream>>>(Wres, Wrt, DIN, DOUT);

    // --- residual GEMM: idr = x @ Wres ---
    {
        dim3 g(N / 32, (DOUT + 511) / 512);
        gemm_bf16_wmma<<<g, 256, 0, stream>>>(xb, Wrt, idr, N, DOUT, DIN);
    }

    // --- layer 1 ---
    {
        dim3 g(N / 32, (H1 + 511) / 512);
        gemm_bf16_wmma<<<g, 256, 0, stream>>>(xb, W1t, hw, N, H1, DIN);
        long long NH = (long long)N * H1;
        k_self_init<<<blocks(NH), T, 0, stream>>>(agg, hw, sn, b1, NH, H1);
        k_scatter<<<E, T, 0, stream>>>(agg, hw, en, src, dst, H1);
        k_bn_relu_bf16<<<blocks(NH), T, 0, stream>>>(agg, hb, g1, be1, m1, v1, NH, H1);
    }

    // --- layer 2 ---
    {
        dim3 g(N / 32, (H2 + 511) / 512);
        gemm_bf16_wmma<<<g, 256, 0, stream>>>(hb, W2t, hw, N, H2, H1);
        long long NH = (long long)N * H2;
        k_self_init<<<blocks(NH), T, 0, stream>>>(agg, hw, sn, b2, NH, H2);
        k_scatter<<<E, T, 0, stream>>>(agg, hw, en, src, dst, H2);
        k_bn_relu_bf16<<<blocks(NH), T, 0, stream>>>(agg, hb, g2, be2, m2, v2, NH, H2);
    }

    // --- layer 3 ---
    {
        dim3 g(N / 32, (DOUT + 511) / 512);
        gemm_bf16_wmma<<<g, 256, 0, stream>>>(hb, W3t, hw, N, DOUT, H2);
        long long NH = (long long)N * DOUT;
        k_self_init<<<blocks(NH), T, 0, stream>>>(agg, hw, sn, b3, NH, DOUT);
        k_scatter<<<E, T, 0, stream>>>(agg, hw, en, src, dst, DOUT);
        k_bn_relu_f32<<<blocks(NH), T, 0, stream>>>(agg, g3, be3, m3, v3, NH, DOUT);
        k_final<<<blocks(NH), T, 0, stream>>>(out, agg, idr, bres, NH, DOUT);
    }
}
